// SelfAttention_33105607918049
// MI455X (gfx1250) — matrix operand order repaired and checked by reading, not compiled, not run on hardware
//
#include <hip/hip_runtime.h>

// ---------------------------------------------------------------------------
// DGCNN-style forward for MI455X (gfx1250, wave32, WMMA).
// All channel matmuls run on v_wmma_f32_16x16x32_f16 (f16 in, f32 accum).
// Conv biases are algebraically dead (bias + affine-free InstanceNorm cancels).
// Problem sizes are fixed by the reference: B=4, C=128, N=4096, K=10.
// GEMM: each wave owns a 64x16 output strip (4 accumulators) so the streamed
// XT operand (the bandwidth-dominant one) is loaded exactly once per K-step.
// ---------------------------------------------------------------------------

#define BB 4
#define CC 128
#define NN 4096
#define KK 10

typedef _Float16 half_t;
typedef __attribute__((ext_vector_type(16))) _Float16 v16h;
typedef __attribute__((ext_vector_type(8)))  _Float16 v8h;
typedef __attribute__((ext_vector_type(8)))  float    v8f;

// ---------------------------------------------------------------------------
// KNN: per query point keep 11 smallest squared distances (self included),
// emit neighbor indices 1..10.  Coords tiled through LDS.
// ---------------------------------------------------------------------------
__global__ void knn_kernel(const float* __restrict__ coords, int* __restrict__ idx)
{
    const int b = blockIdx.y;
    const int n = blockIdx.x * 256 + threadIdx.x;
    const float* cx = coords + (size_t)b * 2 * NN;
    const float* cy = cx + NN;
    __shared__ float sx[256], sy[256];
    float px = cx[n], py = cy[n];

    float bd[KK + 1];
    int   bi[KK + 1];
#pragma unroll
    for (int i = 0; i <= KK; ++i) { bd[i] = 3.4e38f; bi[i] = 0; }

    for (int t = 0; t < NN; t += 256) {
        sx[threadIdx.x] = cx[t + threadIdx.x];
        sy[threadIdx.x] = cy[t + threadIdx.x];
        __syncthreads();
        for (int jj = 0; jj < 256; ++jj) {
            float dx = px - sx[jj], dy = py - sy[jj];
            float d  = fmaxf(dx * dx + dy * dy, 1e-12f);
            if (d < bd[KK]) {
                bd[KK] = d; bi[KK] = t + jj;
#pragma unroll
                for (int p = KK; p > 0; --p) {
                    if (bd[p] < bd[p - 1]) {
                        float td = bd[p]; bd[p] = bd[p - 1]; bd[p - 1] = td;
                        int   ti = bi[p]; bi[p] = bi[p - 1]; bi[p - 1] = ti;
                    }
                }
            }
        }
        __syncthreads();
    }
#pragma unroll
    for (int k = 0; k < KK; ++k)
        idx[((size_t)b * NN + n) * KK + k] = bi[k + 1];   // drop self
}

// ---------------------------------------------------------------------------
// Generic WMMA GEMM:  D[b][M][L] = W[M][Kd] * X[b][L][Kd]^T   (f16 -> f32)
// Each wave computes a 64x16 strip of D: one B fragment per K-step feeds 4
// WMMAs (4 M-tiles).  4 waves per block cover 4 distinct 16-column strips,
// so the streamed XT operand is never fetched redundantly.
// A fragment: lane row = lane&15, halves [hi*16,+8) and [hi*16+8,+8) of each K-strip.
// B fragment: lane col = lane&15, halves [hi*16,+16) contiguous (XT is
// column-major-by-K so this is one 32B load).
// grid = (L/64, M/64, B), block = 128.
// ---------------------------------------------------------------------------
__global__ void wmma_gemm(const half_t* __restrict__ W, const half_t* __restrict__ XT,
                          float* __restrict__ D, int M, int Kd, int L)
{
    const int lane  = threadIdx.x & 31;
    const int wave  = threadIdx.x >> 5;
    const int col0  = (blockIdx.x * 4 + wave) * 16;   // this wave's column strip
    const int mBase = blockIdx.y * 64;                // this block's 64-row band
    const int row   = lane & 15;
    const int hi    = lane >> 4;

    const half_t* __restrict__ Xb = XT + (size_t)blockIdx.z * L * Kd;
    float*        __restrict__ Db = D  + (size_t)blockIdx.z * M * L;

    const half_t* arow = W  + (size_t)(mBase + row) * Kd + hi * 16;
    const half_t* bcol = Xb + (size_t)(col0 + row) * Kd + hi * 16;
    const size_t  mStep = (size_t)16 * Kd;            // next 16-row A tile

    v8f acc0 = {}, acc1 = {}, acc2 = {}, acc3 = {};
    for (int k0 = 0; k0 < Kd; k0 += 32) {
        __builtin_prefetch(bcol + k0 + 32, 0, 1);
        v16h bfrag = *(const v16h*)(bcol + k0);       // K = k0 + hi*16 + [0,16)

        v8h a_lo, a_hi;
        v16h a;
#pragma unroll
        for (int r = 0; r < 4; ++r) {
            a_lo = *(const v8h*)(arow + r * mStep + k0);        // K = k0+hi*16+[0,8)
            a_hi = *(const v8h*)(arow + r * mStep + k0 + 8);    // K = k0+hi*16+[8,16)
#pragma unroll
            for (int i = 0; i < 8; ++i) { a[i] = a_lo[i]; a[8 + i] = a_hi[i]; }
            if      (r == 0) acc0 = __builtin_amdgcn_wmma_f32_16x16x32_f16(false, a, false, bfrag, (short)0, acc0, false, false);
            else if (r == 1) acc1 = __builtin_amdgcn_wmma_f32_16x16x32_f16(false, a, false, bfrag, (short)0, acc1, false, false);
            else if (r == 2) acc2 = __builtin_amdgcn_wmma_f32_16x16x32_f16(false, a, false, bfrag, (short)0, acc2, false, false);
            else             acc3 = __builtin_amdgcn_wmma_f32_16x16x32_f16(false, a, false, bfrag, (short)0, acc3, false, false);
        }
    }
    // D layout per tile: VGPR r -> row +r+hi*8, col col0 + (lane&15)
#pragma unroll
    for (int r = 0; r < 8; ++r) {
        Db[(size_t)(mBase +      r + hi * 8) * L + col0 + row] = acc0[r];
        Db[(size_t)(mBase + 16 + r + hi * 8) * L + col0 + row] = acc1[r];
        Db[(size_t)(mBase + 32 + r + hi * 8) * L + col0 + row] = acc2[r];
        Db[(size_t)(mBase + 48 + r + hi * 8) * L + col0 + row] = acc3[r];
    }
}

// ---------------------------------------------------------------------------
// Im2col / gather builders (all write f16, column-major-by-K rows).
// ---------------------------------------------------------------------------

// graph_feat + width-3/stride-3 im2col:  XT[b][n*3+w][ci*3+t]
__global__ void gather_gf_conv3(const float* __restrict__ S, const int* __restrict__ idx,
                                half_t* __restrict__ XT)
{
    const size_t total = (size_t)3 * NN * 768;
    size_t tid = (size_t)blockIdx.x * 256 + threadIdx.x;
    if (tid >= total) return;
    const int b   = blockIdx.z;
    const int col = (int)(tid / 768);
    const int r   = (int)(tid % 768);
    const int n = col / 3, w = col % 3;
    const int ci = r / 3, t = r % 3;
    const float* Sb = S + (size_t)b * CC * NN;
    float v;
    if (ci < CC) {
        v = Sb[(size_t)ci * NN + n];                          // center feature
    } else {
        const int c  = ci - CC;
        const int kk = w * 3 + t;                             // kk in [0,9)
        const int j  = idx[((size_t)b * NN + n) * KK + kk];
        v = Sb[(size_t)c * NN + j] - Sb[(size_t)c * NN + n];  // neighbor - center
    }
    XT[(size_t)b * total + tid] = (half_t)v;
}

// graph_feat, no conv window:  XT[b][n*10+k][ci], Kd=256
__global__ void gather_gf_plain(const float* __restrict__ S, const int* __restrict__ idx,
                                half_t* __restrict__ XT)
{
    const size_t total = (size_t)NN * KK * 256;
    size_t tid = (size_t)blockIdx.x * 256 + threadIdx.x;
    if (tid >= total) return;
    const int b   = blockIdx.z;
    const int col = (int)(tid >> 8);
    const int ci  = (int)(tid & 255);
    const int n = col / KK, k = col % KK;
    const float* Sb = S + (size_t)b * CC * NN;
    float v;
    if (ci < CC) {
        v = Sb[(size_t)ci * NN + n];
    } else {
        const int c = ci - CC;
        const int j = idx[((size_t)b * NN + n) * KK + k];
        v = Sb[(size_t)c * NN + j] - Sb[(size_t)c * NN + n];
    }
    XT[(size_t)b * total + tid] = (half_t)v;
}

// width-3/stride-3 im2col of Y[b][128][3N]:  XT[b][n][ci*3+t], Kd=384
__global__ void gather_y_conv3(const float* __restrict__ Y, half_t* __restrict__ XT)
{
    const size_t total = (size_t)NN * 384;
    size_t tid = (size_t)blockIdx.x * 256 + threadIdx.x;
    if (tid >= total) return;
    const int b   = blockIdx.z;
    const int col = (int)(tid / 384);
    const int r   = (int)(tid % 384);
    const int ci = r / 3, t = r % 3;
    const float* Yb = Y + (size_t)b * CC * (3 * NN);
    XT[(size_t)b * total + tid] = (half_t)Yb[(size_t)ci * (3 * NN) + col * 3 + t];
}

// cosine-angle + width-3/stride-3 im2col, Kd padded 3 -> 32 with zeros
__global__ void ang_build(const float* __restrict__ coords, const int* __restrict__ idx,
                          half_t* __restrict__ XT)
{
    const size_t total = (size_t)3 * NN * 32;
    size_t tid = (size_t)blockIdx.x * 256 + threadIdx.x;
    if (tid >= total) return;
    const int b   = blockIdx.z;
    const int col = (int)(tid >> 5);
    const int r   = (int)(tid & 31);
    float v = 0.f;
    if (r < 3) {
        const int n = col / 3, w = col % 3;
        const int kk = w * 3 + r;
        const int j  = idx[((size_t)b * NN + n) * KK + kk];
        const float* cx = coords + (size_t)b * 2 * NN;
        const float* cy = cx + NN;
        float ax = cx[n], ay = cy[n], bx = cx[j], by = cy[j];
        float dot = ax * bx + ay * by;
        float na  = sqrtf(ax * ax + ay * ay);
        float nb  = sqrtf(bx * bx + by * by);
        v = dot / (na * nb);
    }
    XT[(size_t)b * total + tid] = (half_t)v;
}

// concat[x0, x1+f_ang, x2+f_ang]:  XT[b][n][r], Kd=384
__global__ void concat3(const float* __restrict__ F, const float* __restrict__ X1,
                        const float* __restrict__ FA, const float* __restrict__ X2,
                        half_t* __restrict__ XT)
{
    const size_t total = (size_t)NN * 384;
    size_t tid = (size_t)blockIdx.x * 256 + threadIdx.x;
    if (tid >= total) return;
    const int b = blockIdx.z;
    const int n = (int)(tid / 384);
    const int r = (int)(tid % 384);
    const size_t base = (size_t)b * CC * NN;
    float v;
    if (r < CC)            v = F [base + (size_t)r * NN + n];
    else if (r < 2 * CC)   v = X1[base + (size_t)(r - CC) * NN + n]
                             + FA[base + (size_t)(r - CC) * NN + n];
    else                   v = X2[base + (size_t)(r - 2 * CC) * NN + n]
                             + FA[base + (size_t)(r - 2 * CC) * NN + n];
    XT[(size_t)b * total + tid] = (half_t)v;
}

// concat[x0, x1o, x2o(256ch)]:  XT[b][n][r], Kd=512
__global__ void concat4(const float* __restrict__ F, const float* __restrict__ X1O,
                        const float* __restrict__ X2O, half_t* __restrict__ XT)
{
    const size_t total = (size_t)NN * 512;
    size_t tid = (size_t)blockIdx.x * 256 + threadIdx.x;
    if (tid >= total) return;
    const int b = blockIdx.z;
    const int n = (int)(tid >> 9);
    const int r = (int)(tid & 511);
    float v;
    if (r < CC)           v = F  [(size_t)b * CC * NN + (size_t)r * NN + n];
    else if (r < 2 * CC)  v = X1O[(size_t)b * CC * NN + (size_t)(r - CC) * NN + n];
    else                  v = X2O[(size_t)b * 2 * CC * NN + (size_t)(r - 2 * CC) * NN + n];
    XT[(size_t)b * total + tid] = (half_t)v;
}

// ---------------------------------------------------------------------------
// InstanceNorm (biased var, eps=1e-5) + activation, in place.
// slope=0 -> ReLU, slope=0.2 -> LeakyReLU.  grid=(M,B), block=256.
// ---------------------------------------------------------------------------
__global__ void instnorm_act(float* __restrict__ X, int M, int L, float slope)
{
    const int b = blockIdx.y, m = blockIdx.x;
    float* p = X + ((size_t)b * M + m) * L;
    float s = 0.f, s2 = 0.f;
    for (int i = threadIdx.x; i < L; i += 256) { float v = p[i]; s += v; s2 += v * v; }
    __shared__ float sh[256], sh2[256];
    sh[threadIdx.x] = s; sh2[threadIdx.x] = s2;
    __syncthreads();
    for (int st = 128; st > 0; st >>= 1) {
        if (threadIdx.x < st) {
            sh[threadIdx.x]  += sh[threadIdx.x + st];
            sh2[threadIdx.x] += sh2[threadIdx.x + st];
        }
        __syncthreads();
    }
    const float mean = sh[0] / (float)L;
    const float var  = sh2[0] / (float)L - mean * mean;
    const float inv  = rsqrtf(var + 1e-5f);
    for (int i = threadIdx.x; i < L; i += 256) {
        float v = (p[i] - mean) * inv;
        p[i] = v > 0.f ? v : v * slope;
    }
}

// max over K: X[b][M][N*10] -> Y[b][M][N]
__global__ void maxk_kernel(const float* __restrict__ X, float* __restrict__ Y, int M)
{
    const size_t total = (size_t)M * NN;
    size_t tid = (size_t)blockIdx.x * 256 + threadIdx.x;
    if (tid >= total) return;
    const int b = blockIdx.z;
    const int m = (int)(tid / NN);
    const int n = (int)(tid % NN);
    const float* p = X + ((size_t)b * M + m) * ((size_t)NN * KK) + (size_t)n * KK;
    float mx = p[0];
#pragma unroll
    for (int k = 1; k < KK; ++k) mx = fmaxf(mx, p[k]);
    Y[((size_t)b * M + m) * NN + n] = mx;
}

__global__ void final_add(const float* __restrict__ A, const float* __restrict__ Bv,
                          float* __restrict__ O, int n)
{
    int i = blockIdx.x * 256 + threadIdx.x;
    if (i < n) O[i] = A[i] + Bv[i];
}

// fp32 -> fp16 weight conversion
__global__ void cvt_f16(const float* __restrict__ s, half_t* __restrict__ d, int n)
{
    int i = blockIdx.x * 256 + threadIdx.x;
    if (i < n) d[i] = (half_t)s[i];
}
// angwA [128][3] -> padded [128][32]
__global__ void cvt_ang(const float* __restrict__ s, half_t* __restrict__ d)
{
    int i = blockIdx.x * 256 + threadIdx.x;
    if (i >= 128 * 32) return;
    int o = i >> 5, r = i & 31;
    d[i] = (r < 3) ? (half_t)s[o * 3 + r] : (half_t)0.f;
}

// ---------------------------------------------------------------------------
// Launch
// ---------------------------------------------------------------------------
extern "C" void kernel_launch(void* const* d_in, const int* in_sizes, int n_in,
                              void* d_out, int out_size, void* d_ws, size_t ws_size,
                              hipStream_t stream)
{
    (void)in_sizes; (void)n_in; (void)out_size; (void)ws_size;
    const float* coords   = (const float*)d_in[0];
    const float* features = (const float*)d_in[1];
    const float* W1  = (const float*)d_in[2];
    const float* W2  = (const float*)d_in[3];
    const float* W3  = (const float*)d_in[4];
    const float* W3o = (const float*)d_in[5];
    const float* a1wA = (const float*)d_in[6];
    const float* a1wB = (const float*)d_in[8];
    const float* a2wA = (const float*)d_in[10];
    const float* a2wB = (const float*)d_in[12];
    const float* angwA = (const float*)d_in[14];
    const float* angwB = (const float*)d_in[16];
    float* out = (float*)d_out;

    // ---- workspace layout ----
    char* ws = (char*)d_ws;
    size_t off = 0;
    auto alloc = [&](size_t bytes) { size_t o = off; off += (bytes + 255) & ~(size_t)255; return o; };
    int*    idx   = (int*)(ws + alloc((size_t)BB * NN * KK * 4));
    half_t* wa1A  = (half_t*)(ws + alloc(128 * 768 * 2));
    half_t* wa1B  = (half_t*)(ws + alloc(128 * 384 * 2));
    half_t* wa2A  = (half_t*)(ws + alloc(128 * 768 * 2));
    half_t* wa2B  = (half_t*)(ws + alloc(128 * 384 * 2));
    half_t* wangA = (half_t*)(ws + alloc(128 * 32 * 2));
    half_t* wangB = (half_t*)(ws + alloc(128 * 384 * 2));
    half_t* w1f   = (half_t*)(ws + alloc(128 * 256 * 2));
    half_t* w2f   = (half_t*)(ws + alloc(256 * 256 * 2));
    half_t* w3f   = (half_t*)(ws + alloc(128 * 384 * 2));
    half_t* w3of  = (half_t*)(ws + alloc(128 * 512 * 2));
    half_t* XT    = (half_t*)(ws + alloc((size_t)BB * NN * KK * 256 * 2));  // 84 MB, reused
    float*  Dbuf  = (float*)(ws + alloc((size_t)BB * 256 * NN * KK * 4));   // 168 MB, reused
    float*  x1    = (float*)(ws + alloc((size_t)BB * CC * NN * 4));
    float*  fang  = (float*)(ws + alloc((size_t)BB * CC * NN * 4));
    float*  x2    = (float*)(ws + alloc((size_t)BB * CC * NN * 4));
    float*  x1o   = (float*)(ws + alloc((size_t)BB * CC * NN * 4));
    float*  x2o   = (float*)(ws + alloc((size_t)BB * 2 * CC * NN * 4));
    float*  x3    = (float*)(ws + alloc((size_t)BB * CC * NN * 4));

    auto nb = [](size_t total) { return (unsigned)((total + 255) / 256); };
    const size_t T_gf3  = (size_t)3 * NN * 768;
    const size_t T_gfp  = (size_t)NN * KK * 256;
    const size_t T_yc3  = (size_t)NN * 384;
    const size_t T_ang  = (size_t)3 * NN * 32;
    const size_t T_c3   = (size_t)NN * 384;
    const size_t T_c4   = (size_t)NN * 512;

    // ---- 0: KNN + weight conversion (conv biases cancel under InstanceNorm) ----
    knn_kernel<<<dim3(NN / 256, BB), 256, 0, stream>>>(coords, idx);
    cvt_f16<<<nb(128 * 768), 256, 0, stream>>>(a1wA, wa1A, 128 * 768);
    cvt_f16<<<nb(128 * 384), 256, 0, stream>>>(a1wB, wa1B, 128 * 384);
    cvt_f16<<<nb(128 * 768), 256, 0, stream>>>(a2wA, wa2A, 128 * 768);
    cvt_f16<<<nb(128 * 384), 256, 0, stream>>>(a2wB, wa2B, 128 * 384);
    cvt_ang<<<nb(128 * 32), 256, 0, stream>>>(angwA, wangA);
    cvt_f16<<<nb(128 * 384), 256, 0, stream>>>(angwB, wangB, 128 * 384);
    cvt_f16<<<nb(128 * 256), 256, 0, stream>>>(W1, w1f, 128 * 256);
    cvt_f16<<<nb(256 * 256), 256, 0, stream>>>(W2, w2f, 256 * 256);
    cvt_f16<<<nb(128 * 384), 256, 0, stream>>>(W3, w3f, 128 * 384);
    cvt_f16<<<nb(128 * 512), 256, 0, stream>>>(W3o, w3of, 128 * 512);

    // ---- 1: a1 block: graph_feat(features) -> conv(768) -> IN/ReLU -> conv(384) -> IN/ReLU ----
    gather_gf_conv3<<<dim3(nb(T_gf3), 1, BB), 256, 0, stream>>>(features, idx, XT);
    wmma_gemm<<<dim3(3 * NN / 64, 2, BB), 128, 0, stream>>>(wa1A, XT, Dbuf, 128, 768, 3 * NN);
    instnorm_act<<<dim3(128, BB), 256, 0, stream>>>(Dbuf, 128, 3 * NN, 0.f);
    gather_y_conv3<<<dim3(nb(T_yc3), 1, BB), 256, 0, stream>>>(Dbuf, XT);
    wmma_gemm<<<dim3(NN / 64, 2, BB), 128, 0, stream>>>(wa1B, XT, x1, 128, 384, NN);
    instnorm_act<<<dim3(128, BB), 256, 0, stream>>>(x1, 128, NN, 0.f);

    // ---- 2: angle encoder (shared between x1 and x2 additions) ----
    ang_build<<<dim3(nb(T_ang), 1, BB), 256, 0, stream>>>(coords, idx, XT);
    wmma_gemm<<<dim3(3 * NN / 64, 2, BB), 128, 0, stream>>>(wangA, XT, Dbuf, 128, 32, 3 * NN);
    instnorm_act<<<dim3(128, BB), 256, 0, stream>>>(Dbuf, 128, 3 * NN, 0.f);
    gather_y_conv3<<<dim3(nb(T_yc3), 1, BB), 256, 0, stream>>>(Dbuf, XT);
    wmma_gemm<<<dim3(NN / 64, 2, BB), 128, 0, stream>>>(wangB, XT, fang, 128, 384, NN);
    instnorm_act<<<dim3(128, BB), 256, 0, stream>>>(fang, 128, NN, 0.f);

    // ---- 3: a2 block on graph_feat(x1) ----
    gather_gf_conv3<<<dim3(nb(T_gf3), 1, BB), 256, 0, stream>>>(x1, idx, XT);
    wmma_gemm<<<dim3(3 * NN / 64, 2, BB), 128, 0, stream>>>(wa2A, XT, Dbuf, 128, 768, 3 * NN);
    instnorm_act<<<dim3(128, BB), 256, 0, stream>>>(Dbuf, 128, 3 * NN, 0.f);
    gather_y_conv3<<<dim3(nb(T_yc3), 1, BB), 256, 0, stream>>>(Dbuf, XT);
    wmma_gemm<<<dim3(NN / 64, 2, BB), 128, 0, stream>>>(wa2B, XT, x2, 128, 384, NN);
    instnorm_act<<<dim3(128, BB), 256, 0, stream>>>(x2, 128, NN, 0.f);

    // ---- 4: W1 path: 1x1 conv over graph_feat(features), IN+lrelu, max over K ----
    gather_gf_plain<<<dim3(nb(T_gfp), 1, BB), 256, 0, stream>>>(features, idx, XT);
    wmma_gemm<<<dim3(NN * KK / 64, 2, BB), 128, 0, stream>>>(w1f, XT, Dbuf, 128, 256, NN * KK);
    instnorm_act<<<dim3(128, BB), 256, 0, stream>>>(Dbuf, 128, NN * KK, 0.2f);
    maxk_kernel<<<dim3(nb((size_t)128 * NN), 1, BB), 256, 0, stream>>>(Dbuf, x1o, 128);

    // ---- 5: W2 path on graph_feat(x1o), 256 output channels ----
    gather_gf_plain<<<dim3(nb(T_gfp), 1, BB), 256, 0, stream>>>(x1o, idx, XT);
    wmma_gemm<<<dim3(NN * KK / 64, 4, BB), 128, 0, stream>>>(w2f, XT, Dbuf, 256, 256, NN * KK);
    instnorm_act<<<dim3(256, BB), 256, 0, stream>>>(Dbuf, 256, NN * KK, 0.2f);
    maxk_kernel<<<dim3(nb((size_t)256 * NN), 1, BB), 256, 0, stream>>>(Dbuf, x2o, 256);

    // ---- 6: W3 on concat[x0, x1+f_ang, x2+f_ang] ----
    concat3<<<dim3(nb(T_c3), 1, BB), 256, 0, stream>>>(features, x1, fang, x2, XT);
    wmma_gemm<<<dim3(NN / 64, 2, BB), 128, 0, stream>>>(w3f, XT, x3, 128, 384, NN);
    instnorm_act<<<dim3(128, BB), 256, 0, stream>>>(x3, 128, NN, 0.2f);

    // ---- 7: W3o on concat[x0, x1o, x2o]; final sum ----
    concat4<<<dim3(nb(T_c4), 1, BB), 256, 0, stream>>>(features, x1o, x2o, XT);
    wmma_gemm<<<dim3(NN / 64, 2, BB), 128, 0, stream>>>(w3of, XT, Dbuf, 128, 512, NN);
    instnorm_act<<<dim3(128, BB), 256, 0, stream>>>(Dbuf, 128, NN, 0.2f);
    final_add<<<nb((size_t)BB * CC * NN), 256, 0, stream>>>(x3, Dbuf, out, BB * CC * NN);
}
